// OmniAttentionMechanism_58652073394282
// MI455X (gfx1250) — matrix operand order repaired and checked by reading, not compiled, not run on hardware
//
#include <hip/hip_runtime.h>

// MI455X / gfx1250 flash-attention with mixed (t2i / lm / mmu) masks.
// B=6 (2 t2i, 2 lm, 2 mmu), H=16, S=1280, D=64, fp32 in/out, bf16 WMMA compute.
// Row-sums on the matrix pipe (all-ones B fragment); fragment loads software-pipelined
// so WMMAs see partial s_wait_dscnt instead of full DS drains.

#define SLEN 1280
#define NH   16
#define NB   6
#define DH   64
#define MMU_END 1027
#define IMG_B 128
#define IMG_E 1152

typedef __attribute__((ext_vector_type(16))) __bf16 v16bf;
typedef __attribute__((ext_vector_type(8)))  __bf16 v8bf;
typedef __attribute__((ext_vector_type(4)))  __bf16 v4bf;
typedef __attribute__((ext_vector_type(8)))  float  v8f;

__device__ __forceinline__ bool mask_ok(int b, int qi, int kv) {
    if (b >= 4) return (qi >= kv) || (kv < MMU_END);       // mmu
    if (b >= 2) return (qi >= kv);                         // lm
    const int pe = (b == 0) ? 80 : 100;                    // t2i pad ends
    const bool causal = ((kv >= pe) && (qi >= kv)) || (qi == kv);
    const bool full   = (qi >= IMG_B) && (qi < IMG_E) && (kv >= IMG_B) && (kv < IMG_E);
    return causal || full;
}

__device__ __forceinline__ v16bf cat16(v8bf lo, v8bf hi) {
    return __builtin_shufflevector(lo, hi, 0,1,2,3,4,5,6,7,8,9,10,11,12,13,14,15);
}

__global__ __launch_bounds__(128)
void omni_flash_attn(const float* __restrict__ Q, const float* __restrict__ K,
                     const float* __restrict__ V, float* __restrict__ O)
{
    __shared__ __bf16 Kt[64][72];        // K tile, row-major [key][d]; pad keeps 16B align
    __shared__ __bf16 Vt[64][72];        // V tile, transposed [d][key]
    __shared__ __bf16 Pt[4][16][72];     // per-wave probability bounce buffer

    const int tid  = threadIdx.x;
    const int wave = tid >> 5;
    const int lane = tid & 31;
    const int col  = lane & 15;
    const int half = lane >> 4;

    const int q0 = blockIdx.x * 64;
    const int bh = blockIdx.y;
    const int b  = bh / NH;
    const size_t base = (size_t)bh * SLEN * DH;

    // ---- Q fragments (A layout, 16x32 bf16 per chunk), softmax scale folded in ----
    const int qrowA = q0 + wave * 16 + col;          // A layout: lane holds row m = lane%16
    const float* qp = Q + base + (size_t)qrowA * DH;
    v16bf aQ[2];
#pragma unroll
    for (int c = 0; c < 2; ++c) {
        const int d0 = c * 32 + half * 8;            // elements 0..7  -> K = d0 + e
        const int d1 = d0 + 16;                      // elements 8..15 -> K = d1 + e
        float4 f0 = *(const float4*)(qp + d0);
        float4 f1 = *(const float4*)(qp + d0 + 4);
        float4 f2 = *(const float4*)(qp + d1);
        float4 f3 = *(const float4*)(qp + d1 + 4);
        float tmp[16] = {f0.x,f0.y,f0.z,f0.w, f1.x,f1.y,f1.z,f1.w,
                         f2.x,f2.y,f2.z,f2.w, f3.x,f3.y,f3.z,f3.w};
#pragma unroll
        for (int e = 0; e < 16; ++e) aQ[c][e] = (__bf16)(tmp[e] * 0.125f);
    }

    // all-ones B fragment: every column of (P x ones) is rowsum(P)
    v16bf bones;
#pragma unroll
    for (int e = 0; e < 16; ++e) bones[e] = (__bf16)1.0f;

    // ---- flash state: C-layout rows are (half*8 + v) ----
    float mrow[8];
    v8f Oc[4], Lc;
#pragma unroll
    for (int v = 0; v < 8; ++v) { mrow[v] = -1e30f; Lc[v] = 0.0f; }
#pragma unroll
    for (int g = 0; g < 4; ++g)
#pragma unroll
        for (int e = 0; e < 8; ++e) Oc[g][e] = 0.0f;

    const int qmax  = q0 + 63;
    const int qwmin = q0 + wave * 16;                // wave's q-row range
    const int qwmax = qwmin + 15;
    const int qrowC = qwmin + half * 8;

    for (int key0 = 0; key0 < SLEN; key0 += 64) {
        // -------- block-uniform whole-tile skip --------
        bool need = (key0 <= qmax);
        if (!need) {
            if (b < 2)       need = (q0 < IMG_E) && (qmax >= IMG_B) && (key0 < IMG_E);
            else if (b >= 4) need = (key0 < MMU_END);
        }
        if (!need) continue;

        __syncthreads();   // previous tile fully consumed

        // -------- stage K (row-major) and V (transposed) as bf16 --------
#pragma unroll
        for (int i = 0; i < 8; ++i) {
            const int linear = i * 512 + tid * 4;
            const int r  = linear >> 6;
            const int cc = linear & 63;
            const float4 k4 = *(const float4*)(K + base + (size_t)(key0 + r) * DH + cc);
            const float4 v4 = *(const float4*)(V + base + (size_t)(key0 + r) * DH + cc);
            v4bf kb = {(__bf16)k4.x, (__bf16)k4.y, (__bf16)k4.z, (__bf16)k4.w};
            *(v4bf*)&Kt[r][cc] = kb;
            Vt[cc + 0][r] = (__bf16)v4.x;
            Vt[cc + 1][r] = (__bf16)v4.y;
            Vt[cc + 2][r] = (__bf16)v4.z;
            Vt[cc + 3][r] = (__bf16)v4.w;
        }
        if (key0 + 64 < SLEN) {   // L2 prefetch of next tile -> global_prefetch_b8
            __builtin_prefetch(K + base + (size_t)(key0 + 64) * DH + tid * 32, 0, 1);
            __builtin_prefetch(V + base + (size_t)(key0 + 64) * DH + tid * 32, 0, 1);
        }
        __syncthreads();

        // -------- wave-uniform skip: this wave's rows see nothing in this tile --------
        bool wneed = (key0 <= qwmax);
        if (!wneed) {
            if (b < 2)       wneed = (qwmin < IMG_E) && (qwmax >= IMG_B) && (key0 < IMG_E);
            else if (b >= 4) wneed = (key0 < MMU_END);
        }
        if (!wneed) continue;

        // -------- preload all 8 K^T fragments, then WMMA chain (pipelined DS) --------
        v16bf bK[4][2];
#pragma unroll
        for (int j = 0; j < 4; ++j)
#pragma unroll
            for (int c = 0; c < 2; ++c) {
                const __bf16* kp8 = &Kt[j * 16 + col][c * 32 + half * 8];
                bK[j][c] = cat16(*(const v8bf*)kp8, *(const v8bf*)(kp8 + 16));
            }

        v8f Sc[4];
#pragma unroll
        for (int j = 0; j < 4; ++j) {
#pragma unroll
            for (int e = 0; e < 8; ++e) Sc[j][e] = 0.0f;
#pragma unroll
            for (int c = 0; c < 2; ++c)
                Sc[j] = __builtin_amdgcn_wmma_f32_16x16x32_bf16(
                            false, aQ[c], false, bK[j][c], (short)0, Sc[j], false, false);
        }

        // -------- masking: only boundary tiles pay --------
        bool tile_full = (key0 + 63 <= qwmin);       // strictly below diagonal
        if (!tile_full) {
            if (b >= 4)      tile_full = (key0 + 63 < MMU_END);
            else if (b < 2)  tile_full = (qwmin >= IMG_B) && (qwmax < IMG_E) &&
                                         (key0 >= IMG_B) && (key0 + 63 < IMG_E);
        }
        if (!tile_full) {
#pragma unroll
            for (int j = 0; j < 4; ++j) {
                const int kv = key0 + j * 16 + col;
#pragma unroll
                for (int v = 0; v < 8; ++v)
                    Sc[j][v] = mask_ok(b, qrowC + v, kv) ? Sc[j][v] : -1e30f;
            }
        }

        // -------- online softmax: max butterfly only (sum comes from WMMA) --------
        float alpha[8];
#pragma unroll
        for (int v = 0; v < 8; ++v) {
            float tm = fmaxf(fmaxf(Sc[0][v], Sc[1][v]), fmaxf(Sc[2][v], Sc[3][v]));
#pragma unroll
            for (int off = 1; off < 16; off <<= 1)
                tm = fmaxf(tm, __shfl_xor(tm, off, 16));
            const float mn = fmaxf(mrow[v], tm);
            alpha[v] = __expf(mrow[v] - mn);
            mrow[v]  = mn;
        }

        // exponentials -> P (bf16) into LDS (C -> A layout bounce), then rescale O/L
#pragma unroll
        for (int v = 0; v < 8; ++v)
#pragma unroll
            for (int j = 0; j < 4; ++j) {
                const float p = __expf(Sc[j][v] - mrow[v]);
                Pt[wave][half * 8 + v][j * 16 + col] = (__bf16)p;
            }
#pragma unroll
        for (int v = 0; v < 8; ++v) {
            Lc[v] *= alpha[v];
#pragma unroll
            for (int g = 0; g < 4; ++g) Oc[g][v] *= alpha[v];
        }

        // -------- preload ALL GEMM2 fragments (queue behind P stores in-order) --------
        v16bf aP[2], bV[2][4];
#pragma unroll
        for (int kc = 0; kc < 2; ++kc) {
            const __bf16* p0 = &Pt[wave][col][kc * 32 + half * 8];
            const __bf16* p1 = &Pt[wave][col][kc * 32 + 16 + half * 8];
            aP[kc] = cat16(*(const v8bf*)p0, *(const v8bf*)p1);
#pragma unroll
            for (int g = 0; g < 4; ++g) {
                const __bf16* vv = &Vt[g * 16 + col][kc * 32 + half * 8];
                bV[kc][g] = cat16(*(const v8bf*)vv, *(const v8bf*)(vv + 16));
            }
        }

        // -------- O += P * V ; L += P * ones --------
#pragma unroll
        for (int kc = 0; kc < 2; ++kc) {
#pragma unroll
            for (int g = 0; g < 4; ++g)
                Oc[g] = __builtin_amdgcn_wmma_f32_16x16x32_bf16(
                            false, aP[kc], false, bV[kc][g], (short)0, Oc[g], false, false);
            Lc = __builtin_amdgcn_wmma_f32_16x16x32_bf16(
                     false, aP[kc], false, bones, (short)0, Lc, false, false);
        }
    }

    // -------- epilogue: O / L --------
    float* op = O + base;
#pragma unroll
    for (int v = 0; v < 8; ++v) {
        const float inv = 1.0f / Lc[v];
#pragma unroll
        for (int g = 0; g < 4; ++g)
            op[(size_t)(qrowC + v) * DH + g * 16 + col] = Oc[g][v] * inv;
    }
}

extern "C" void kernel_launch(void* const* d_in, const int* in_sizes, int n_in,
                              void* d_out, int out_size, void* d_ws, size_t ws_size,
                              hipStream_t stream) {
    const float* q = (const float*)d_in[0];
    const float* k = (const float*)d_in[1];
    const float* v = (const float*)d_in[2];
    float* o = (float*)d_out;
    dim3 grid(SLEN / 64, NB * NH);
    dim3 block(128);
    hipLaunchKernelGGL(omni_flash_attn, grid, block, 0, stream, q, k, v, o);
}
